// DeformableSelfAttention3D_52871047414101
// MI455X (gfx1250) — matrix operand order repaired and checked by reading, not compiled, not run on hardware
//
#include <hip/hip_runtime.h>
#include <hip/hip_bf16.h>
#include <cstdint>

typedef __attribute__((ext_vector_type(16))) _Float16 v16h;
typedef __attribute__((ext_vector_type(8)))  float    v8f;
typedef __attribute__((ext_vector_type(2)))  _Float16 h2;

namespace {
constexpr int B_  = 2;
constexpr int DHW = 20;      // full spatial
constexpr int C_  = 384;
constexpr int N_  = 8000;    // tokens per batch
constexpr int NH  = 8;
constexpr int HD  = 48;
constexpr int DR  = 10;      // reduced spatial
constexpr int M_  = 1000;    // reduced tokens per batch
constexpr int MP  = 1024;    // keys padded to multiple of 32
constexpr int HDP = 64;      // head dim padded to multiple of 32
constexpr int BN  = B_ * N_; // 16000 rows
constexpr int BM  = B_ * M_; // 2000 rows
constexpr int KSR = C_ * 8;  // 3072 (2x2x2 patch * C)
}

// ---------------- WMMA helpers (CDNA5 16x16x32 f16 -> f32) ----------------

__device__ __forceinline__ v8f wmma_f16(v16h a, v16h b, v8f c) {
  return __builtin_amdgcn_wmma_f32_16x16x32_f16(false, a, false, b, (short)0, c,
                                                false, false);
}

// A fragment 16x32 f16. Per ISA 7.12.2: lane L -> m = L&15; k base = (L>=16)?8:0;
// VGPR g holds k = ((g>=4)?16:0) + kbase + 2*(g&3) + {0,1}.
// p must already point at element (m_lane, k0 + kbase_lane).
__device__ __forceinline__ v16h load_a_frag_p(const _Float16* p) {
  v16h a;
#pragma unroll
  for (int g = 0; g < 8; ++g) {
    int k = ((g & 4) << 2) + ((g & 3) << 1);   // 0,2,4,6,16,18,20,22
    h2 t = *(const h2*)(p + k);
    a[2 * g] = t[0];
    a[2 * g + 1] = t[1];
  }
  return a;
}

__device__ __forceinline__ v16h load_a_frag(const _Float16* src, int ld) {
  int lane = threadIdx.x & 31;
  return load_a_frag_p(src + (size_t)(lane & 15) * ld + ((lane >> 4) << 3));
}

// B fragment 32x16 f16 from storage where B[k][n] lives at src[n*ld + k]
// (consecutive k contiguous). lane L -> n = L&15; k base = (L>=16)?8:0;
// VGPR g holds k = ((g>=4)?16:0) + kbase + 2*(g&3) + {0,1}.
__device__ __forceinline__ v16h load_b_frag(const _Float16* src, int ld) {
  int lane = threadIdx.x & 31;
  const _Float16* p = src + (size_t)(lane & 15) * ld + ((lane >> 4) << 3);
  v16h b;
#pragma unroll
  for (int g = 0; g < 8; ++g) {
    h2 t = *(const h2*)(p + ((g & 4) << 2) + ((g & 3) << 1));
    b[2 * g] = t[0];
    b[2 * g + 1] = t[1];
  }
  return b;
}

// ---------------- utility kernels ----------------

__global__ void k_cvt(const float* __restrict__ s, _Float16* __restrict__ d, int n) {
  int i = blockIdx.x * blockDim.x + threadIdx.x;
  if (i < n) d[i] = (_Float16)s[i];
}

__global__ void k_zero32(uint32_t* __restrict__ p, int n) {
  int i = blockIdx.x * blockDim.x + threadIdx.x;
  if (i < n) p[i] = 0u;
}

// Repack sr_w [co][ci][2][2][2] (fp32) -> W[co][k] f16 with k = tap*C + ci,
// tap = dz*4 + dy*2 + dx.
__global__ void k_pack_sr(const float* __restrict__ sw, _Float16* __restrict__ d, int n) {
  int i = blockIdx.x * blockDim.x + threadIdx.x;
  if (i >= n) return;
  int co = i / KSR, k = i % KSR;
  int tap = k / C_, ci = k % C_;
  d[i] = (_Float16)sw[(size_t)(co * C_ + ci) * 8 + tap];
}

// ---------------- q projection: q = x @ q_w^T, stored [b][h][n][64] f16 ----

__global__ void __launch_bounds__(256) k_gemm_q(const _Float16* __restrict__ xh,
                                                const _Float16* __restrict__ qwh,
                                                _Float16* __restrict__ qh) {
  int w = (blockIdx.x * blockDim.x + threadIdx.x) >> 5;
  if (w >= 1000 * NH) return;
  int rt = w % 1000, head = w / 1000;
  int lane = threadIdx.x & 31;
  v8f acc[3] = {};
  const _Float16* arow = xh + (size_t)rt * 16 * C_;
  for (int kk = 0; kk < C_; kk += 32) {
    v16h a = load_a_frag(arow + kk, C_);
#pragma unroll
    for (int t = 0; t < 3; ++t) {
      v16h b = load_b_frag(qwh + (size_t)(head * HD + t * 16) * C_ + kk, C_);
      acc[t] = wmma_f16(a, b, acc[t]);
    }
  }
  int nloc = lane & 15, mhi = (lane >> 4) << 3;
#pragma unroll
  for (int t = 0; t < 3; ++t)
#pragma unroll
    for (int r = 0; r < 8; ++r) {
      int row = rt * 16 + r + mhi;
      int b = row / N_, n = row % N_;
      qh[((size_t)(b * NH + head) * N_ + n) * HDP + t * 16 + nloc] =
          (_Float16)acc[t][r];
    }
}

// ---- stride-2 2^3 conv as GEMM: rows = output voxels, K = 3072, cols = C ----

__global__ void __launch_bounds__(256) k_gemm_sr(const _Float16* __restrict__ xh,
                                                 const _Float16* __restrict__ wsr,
                                                 const float* __restrict__ srb,
                                                 float* __restrict__ xs) {
  int w = (blockIdx.x * blockDim.x + threadIdx.x) >> 5;
  if (w >= 125 * 6) return;
  int rt = w % 125, ct = w / 125;
  int c0 = ct * 64;
  int lane = threadIdx.x & 31;
  int mrow = rt * 16 + (lane & 15);
  int b = mrow / M_, vox = mrow % M_;
  int z = vox / 100, y = (vox / 10) % 10, x = vox % 10;
  int kb = (lane >> 4) << 3;
  v8f acc[4] = {};
  for (int kk = 0; kk < KSR; kk += 32) {
    int tap = kk / C_, ci = kk % C_;
    int dz = tap >> 2, dy = (tap >> 1) & 1, dx = tap & 1;
    int nin = ((2 * z + dz) * DHW + (2 * y + dy)) * DHW + (2 * x + dx);
    v16h a = load_a_frag_p(xh + ((size_t)b * N_ + nin) * C_ + ci + kb);
#pragma unroll
    for (int t = 0; t < 4; ++t) {
      v16h bf = load_b_frag(wsr + (size_t)(c0 + t * 16) * KSR + kk, KSR);
      acc[t] = wmma_f16(a, bf, acc[t]);
    }
  }
  int nloc = lane & 15, mhi = (lane >> 4) << 3;
#pragma unroll
  for (int t = 0; t < 4; ++t)
#pragma unroll
    for (int r = 0; r < 8; ++r) {
      int row = rt * 16 + r + mhi;
      int co = c0 + t * 16 + nloc;
      xs[(size_t)row * C_ + co] = acc[t][r] + srb[co];
    }
}

// ---------------- layernorm over C (one wave per row, in place) ----------------

__global__ void __launch_bounds__(256) k_ln(float* __restrict__ xs,
                                            const float* __restrict__ g,
                                            const float* __restrict__ bt,
                                            _Float16* __restrict__ xh2) {
  int w = (blockIdx.x * blockDim.x + threadIdx.x) >> 5;
  if (w >= BM) return;
  int lane = threadIdx.x & 31;
  float v[12];
  float s = 0.f, s2 = 0.f;
  float* row = xs + (size_t)w * C_;
#pragma unroll
  for (int i = 0; i < 12; ++i) {
    v[i] = row[lane + 32 * i];
    s += v[i];
    s2 += v[i] * v[i];
  }
#pragma unroll
  for (int m = 1; m < 32; m <<= 1) {
    s += __shfl_xor(s, m);
    s2 += __shfl_xor(s2, m);
  }
  float mean = s * (1.f / C_);
  float var = s2 * (1.f / C_) - mean * mean;
  float rs = rsqrtf(var + 1e-5f);
#pragma unroll
  for (int i = 0; i < 12; ++i) {
    int c = lane + 32 * i;
    float o = (v[i] - mean) * rs * g[c] + bt[c];
    row[c] = o;
    xh2[(size_t)w * C_ + c] = (_Float16)o;
  }
}

// ---------------- kv = x_ @ kv_w^T, scattered into k/v [bh][m][48] fp32 ------

__global__ void __launch_bounds__(256) k_gemm_kv(const _Float16* __restrict__ xh2,
                                                 const _Float16* __restrict__ kvwh,
                                                 float* __restrict__ kf,
                                                 float* __restrict__ vf) {
  int w = (blockIdx.x * blockDim.x + threadIdx.x) >> 5;
  if (w >= 125 * 12) return;
  int rt = w % 125, ct = w / 125;
  int c0 = ct * 64;
  int lane = threadIdx.x & 31;
  v8f acc[4] = {};
  const _Float16* arow = xh2 + (size_t)rt * 16 * C_;
  for (int kk = 0; kk < C_; kk += 32) {
    v16h a = load_a_frag(arow + kk, C_);
#pragma unroll
    for (int t = 0; t < 4; ++t) {
      v16h b = load_b_frag(kvwh + (size_t)(c0 + t * 16) * C_ + kk, C_);
      acc[t] = wmma_f16(a, b, acc[t]);
    }
  }
  int nloc = lane & 15, mhi = (lane >> 4) << 3;
#pragma unroll
  for (int t = 0; t < 4; ++t)
#pragma unroll
    for (int r = 0; r < 8; ++r) {
      int row = rt * 16 + r + mhi;
      int b = row / M_, m = row % M_;
      int c = c0 + t * 16 + nloc;
      float val = acc[t][r];
      if (c < C_)
        kf[((size_t)(b * NH + c / HD) * M_ + m) * HD + (c % HD)] = val;
      else {
        int c2 = c - C_;
        vf[((size_t)(b * NH + c2 / HD) * M_ + m) * HD + (c2 % HD)] = val;
      }
    }
}

// ------- 3x3x3 offset conv + tanh*2 + base grid -> samp [bh][p][3] fp32 ------

__global__ void k_offsamp(const float* __restrict__ x_,
                          const float* __restrict__ offw,
                          const float* __restrict__ offb,
                          float* __restrict__ samp) {
  int i = blockIdx.x * blockDim.x + threadIdx.x;
  if (i >= B_ * 24 * M_) return;
  int vox = i % M_;
  int t = i / M_;
  int oc = t % 24, b = t / 24;
  int z = vox / 100, y = (vox / 10) % 10, x = vox % 10;
  float acc = offb[oc];
  for (int dz = -1; dz <= 1; ++dz) {
    int zz = z + dz;
    if (zz < 0 || zz >= DR) continue;
    for (int dy = -1; dy <= 1; ++dy) {
      int yy = y + dy;
      if (yy < 0 || yy >= DR) continue;
      for (int dx = -1; dx <= 1; ++dx) {
        int xx = x + dx;
        if (xx < 0 || xx >= DR) continue;
        int tap = ((dz + 1) * 3 + (dy + 1)) * 3 + (dx + 1);
        const float* xin = x_ + ((size_t)b * M_ + (zz * 10 + yy) * 10 + xx) * C_;
        const float* wr = offw + (size_t)oc * C_ * 27 + tap;
        float a2 = 0.f;
        for (int c = 0; c < C_; ++c) a2 += xin[c] * wr[(size_t)c * 27];
        acc += a2;
      }
    }
  }
  float off = tanhf(acc) * 2.0f;
  int comp = oc % 3, h = oc / 3;
  float bc = (float)(comp == 0 ? z : (comp == 1 ? y : x)) * (2.0f / 9.0f) - 1.0f;
  samp[((size_t)(b * NH + h) * M_ + vox) * 3 + comp] = bc + off;
}

// ------- trilinear sample of k,v -> sk [bh][1024][64] f16, sv^T [bh][48][1024]

__global__ void k_sample(const float* __restrict__ kf, const float* __restrict__ vf,
                         const float* __restrict__ samp,
                         _Float16* __restrict__ skh, _Float16* __restrict__ svt) {
  int i = blockIdx.x * blockDim.x + threadIdx.x;
  if (i >= 16 * M_ * HD) return;
  int d = i % HD;
  int t = i / HD;
  int p = t % M_, bh = t / M_;
  const float* g = samp + ((size_t)bh * M_ + p) * 3;
  // grid_sample semantics: comp0 -> x(W), comp1 -> y(H), comp2 -> z(D)
  float ix = (g[0] + 1.f) * 0.5f * 9.f;
  float iy = (g[1] + 1.f) * 0.5f * 9.f;
  float iz = (g[2] + 1.f) * 0.5f * 9.f;
  float fx0 = floorf(ix), fy0 = floorf(iy), fz0 = floorf(iz);
  float fx = ix - fx0, fy = iy - fy0, fz = iz - fz0;
  int x0 = (int)fx0, y0 = (int)fy0, z0 = (int)fz0;
  float sk = 0.f, sv = 0.f;
#pragma unroll
  for (int dz = 0; dz < 2; ++dz) {
    int zi = z0 + dz;
    float wz = dz ? fz : 1.f - fz;
#pragma unroll
    for (int dy = 0; dy < 2; ++dy) {
      int yi = y0 + dy;
      float wy = dy ? fy : 1.f - fy;
#pragma unroll
      for (int dx = 0; dx < 2; ++dx) {
        int xi = x0 + dx;
        float wx = dx ? fx : 1.f - fx;
        bool valid = (xi >= 0) & (xi < DR) & (yi >= 0) & (yi < DR) &
                     (zi >= 0) & (zi < DR);
        int zc = min(max(zi, 0), DR - 1);
        int yc = min(max(yi, 0), DR - 1);
        int xc = min(max(xi, 0), DR - 1);
        int idx = (zc * 10 + yc) * 10 + xc;
        float wgt = valid ? wx * wy * wz : 0.f;
        sk += kf[((size_t)bh * M_ + idx) * HD + d] * wgt;
        sv += vf[((size_t)bh * M_ + idx) * HD + d] * wgt;
      }
    }
  }
  skh[((size_t)bh * MP + p) * HDP + d] = (_Float16)sk;
  svt[((size_t)bh * HD + d) * MP + p] = (_Float16)sv;
}

// ------- flash attention: one wave per (bh, 16-row q tile), keys 0..1023 -----

__global__ void __launch_bounds__(256) k_attn(const _Float16* __restrict__ qh,
                                              const _Float16* __restrict__ skh,
                                              const _Float16* __restrict__ svt,
                                              _Float16* __restrict__ oh) {
  __shared__ _Float16 P[8][16 * 32];
  int wid = threadIdx.x >> 5;
  int w = blockIdx.x * 8 + wid;
  if (w >= 16 * 500) return;
  int bh = w / 500, qt = w % 500;
  int lane = threadIdx.x & 31;
  int b = bh >> 3, h = bh & 7;
  const _Float16* qbase = qh + ((size_t)bh * N_ + qt * 16) * HDP;
  v16h aq0 = load_a_frag(qbase, HDP);
  v16h aq1 = load_a_frag(qbase + 32, HDP);
  v8f zero = {};
  v8f o0 = zero, o1 = zero, o2 = zero;
  float rmax[8], rsum[8];
#pragma unroll
  for (int r = 0; r < 8; ++r) { rmax[r] = -1e30f; rsum[r] = 0.f; }
  const float scale = 0.14433756729740643f;  // 48^-0.5
  const _Float16* skb = skh + (size_t)bh * MP * HDP;
  const _Float16* svb = svt + (size_t)bh * HD * MP;
  _Float16* pl = &P[wid][0];
  int nloc = lane & 15, mhi = (lane >> 4) << 3;
#pragma unroll 1
  for (int jp = 0; jp < MP / 32; ++jp) {
#pragma unroll
    for (int half = 0; half < 2; ++half) {
      int key0 = jp * 32 + half * 16;
      v16h bk0 = load_b_frag(skb + (size_t)key0 * HDP, HDP);
      v16h bk1 = load_b_frag(skb + (size_t)key0 * HDP + 32, HDP);
      v8f s = wmma_f16(aq0, bk0, zero);
      s = wmma_f16(aq1, bk1, s);
      float fmask = (key0 + nloc < M_) ? 0.f : -1e30f;
      float tm[8];
#pragma unroll
      for (int r = 0; r < 8; ++r) { s[r] = s[r] * scale + fmask; tm[r] = s[r]; }
#pragma unroll
      for (int mx = 1; mx < 16; mx <<= 1)
#pragma unroll
        for (int r = 0; r < 8; ++r) tm[r] = fmaxf(tm[r], __shfl_xor(tm[r], mx));
      float pr[8];
#pragma unroll
      for (int r = 0; r < 8; ++r) {
        float nm = fmaxf(rmax[r], tm[r]);
        float f = __expf(rmax[r] - nm);
        rmax[r] = nm;
        o0[r] *= f; o1[r] *= f; o2[r] *= f;
        rsum[r] *= f;
        pr[r] = __expf(s[r] - nm);
      }
      float ts[8];
#pragma unroll
      for (int r = 0; r < 8; ++r) ts[r] = pr[r];
#pragma unroll
      for (int mx = 1; mx < 16; mx <<= 1)
#pragma unroll
        for (int r = 0; r < 8; ++r) ts[r] += __shfl_xor(ts[r], mx);
#pragma unroll
      for (int r = 0; r < 8; ++r) {
        rsum[r] += ts[r];
        pl[(r + mhi) * 32 + half * 16 + nloc] = (_Float16)pr[r];
      }
    }
    // DS ops are in-order per wave; the wait+clobber stops compile-time
    // reordering across the cross-lane LDS handoff.
    asm volatile("s_wait_dscnt 0" ::: "memory");
    v16h ap = load_a_frag(pl, 32);
    o0 = wmma_f16(ap, load_b_frag(svb + (size_t)0 * 16 * MP + jp * 32, MP), o0);
    o1 = wmma_f16(ap, load_b_frag(svb + (size_t)1 * 16 * MP + jp * 32, MP), o1);
    o2 = wmma_f16(ap, load_b_frag(svb + (size_t)2 * 16 * MP + jp * 32, MP), o2);
  }
#pragma unroll
  for (int r = 0; r < 8; ++r) {
    float inv = 1.f / rsum[r];
    int row = qt * 16 + r + mhi;
    size_t ob = ((size_t)b * N_ + row) * C_ + h * HD;
    oh[ob + 0 + nloc] = (_Float16)(o0[r] * inv);
    oh[ob + 16 + nloc] = (_Float16)(o1[r] * inv);
    oh[ob + 32 + nloc] = (_Float16)(o2[r] * inv);
  }
}

// ---------------- out = attn_out @ proj_w^T + proj_b (fp32 out) ----------------

__global__ void __launch_bounds__(256) k_gemm_proj(const _Float16* __restrict__ oh,
                                                   const _Float16* __restrict__ pwh,
                                                   const float* __restrict__ pb,
                                                   float* __restrict__ out) {
  int w = (blockIdx.x * blockDim.x + threadIdx.x) >> 5;
  if (w >= 1000 * 6) return;
  int rt = w % 1000, ct = w / 1000;
  int c0 = ct * 64;
  int lane = threadIdx.x & 31;
  v8f acc[4] = {};
  const _Float16* arow = oh + (size_t)rt * 16 * C_;
  for (int kk = 0; kk < C_; kk += 32) {
    v16h a = load_a_frag(arow + kk, C_);
#pragma unroll
    for (int t = 0; t < 4; ++t) {
      v16h b = load_b_frag(pwh + (size_t)(c0 + t * 16) * C_ + kk, C_);
      acc[t] = wmma_f16(a, b, acc[t]);
    }
  }
  int nloc = lane & 15, mhi = (lane >> 4) << 3;
#pragma unroll
  for (int t = 0; t < 4; ++t)
#pragma unroll
    for (int r = 0; r < 8; ++r) {
      int row = rt * 16 + r + mhi;
      int c = c0 + t * 16 + nloc;
      out[(size_t)row * C_ + c] = acc[t][r] + pb[c];
    }
}

// ---------------- host launch ----------------

extern "C" void kernel_launch(void* const* d_in, const int* in_sizes, int n_in,
                              void* d_out, int out_size, void* d_ws, size_t ws_size,
                              hipStream_t stream) {
  (void)in_sizes; (void)n_in; (void)out_size; (void)ws_size;
  const float* x      = (const float*)d_in[0];
  const float* q_w    = (const float*)d_in[1];
  const float* kv_w   = (const float*)d_in[2];
  const float* proj_w = (const float*)d_in[3];
  const float* proj_b = (const float*)d_in[4];
  const float* sr_w   = (const float*)d_in[5];
  const float* sr_b   = (const float*)d_in[6];
  const float* ln_g   = (const float*)d_in[7];
  const float* ln_b   = (const float*)d_in[8];
  const float* off_w  = (const float*)d_in[9];
  const float* off_b  = (const float*)d_in[10];
  float* out = (float*)d_out;

  char* base = (char*)d_ws;
  size_t off = 0;
  auto take = [&](size_t bytes) -> char* {
    char* r = base + off;
    off = (off + bytes + 255) & ~(size_t)255;
    return r;
  };
  _Float16* xh   = (_Float16*)take((size_t)BN * C_ * 2);   // x in f16
  _Float16* oh   = xh;                                     // alias: attn out (last x reader runs first)
  _Float16* qh   = (_Float16*)take((size_t)16 * N_ * HDP * 2);
  float*    xs   = (float*)take((size_t)BM * C_ * 4);      // conv out -> layernormed in place
  _Float16* xh2  = (_Float16*)take((size_t)BM * C_ * 2);
  float*    kf   = (float*)take((size_t)16 * M_ * HD * 4);
  float*    vf   = (float*)take((size_t)16 * M_ * HD * 4);
  float*    samp = (float*)take((size_t)16 * M_ * 3 * 4);
  _Float16* skh  = (_Float16*)take((size_t)16 * MP * HDP * 2);
  _Float16* svt  = (_Float16*)take((size_t)16 * HD * MP * 2);
  _Float16* qwh  = (_Float16*)take((size_t)C_ * C_ * 2);
  _Float16* kvwh = (_Float16*)take((size_t)2 * C_ * C_ * 2);
  _Float16* pwh  = (_Float16*)take((size_t)C_ * C_ * 2);
  _Float16* wsr  = (_Float16*)take((size_t)KSR * C_ * 2);

  auto cdiv = [](int a, int b) { return (a + b - 1) / b; };

  k_cvt<<<cdiv(BN * C_, 256), 256, 0, stream>>>(x, xh, BN * C_);
  k_cvt<<<cdiv(C_ * C_, 256), 256, 0, stream>>>(q_w, qwh, C_ * C_);
  k_cvt<<<cdiv(2 * C_ * C_, 256), 256, 0, stream>>>(kv_w, kvwh, 2 * C_ * C_);
  k_cvt<<<cdiv(C_ * C_, 256), 256, 0, stream>>>(proj_w, pwh, C_ * C_);
  k_pack_sr<<<cdiv(C_ * KSR, 256), 256, 0, stream>>>(sr_w, wsr, C_ * KSR);
  k_zero32<<<cdiv(16 * N_ * HDP / 2, 256), 256, 0, stream>>>((uint32_t*)qh, 16 * N_ * HDP / 2);
  k_zero32<<<cdiv(16 * MP * HDP / 2, 256), 256, 0, stream>>>((uint32_t*)skh, 16 * MP * HDP / 2);
  k_zero32<<<cdiv(16 * HD * MP / 2, 256), 256, 0, stream>>>((uint32_t*)svt, 16 * HD * MP / 2);

  k_gemm_q<<<1000, 256, 0, stream>>>(xh, qwh, qh);                     // 8000 waves
  k_gemm_sr<<<94, 256, 0, stream>>>(xh, wsr, sr_b, xs);                // 750 waves
  k_ln<<<250, 256, 0, stream>>>(xs, ln_g, ln_b, xh2);                  // 2000 waves
  k_gemm_kv<<<188, 256, 0, stream>>>(xh2, kvwh, kf, vf);               // 1500 waves
  k_offsamp<<<cdiv(B_ * 24 * M_, 256), 256, 0, stream>>>(xs, off_w, off_b, samp);
  k_sample<<<cdiv(16 * M_ * HD, 256), 256, 0, stream>>>(kf, vf, samp, skh, svt);
  k_attn<<<1000, 256, 0, stream>>>(qh, skh, svt, oh);                  // 8000 waves
  k_gemm_proj<<<750, 256, 0, stream>>>(oh, pwh, proj_b, out);          // 6000 waves
}
